// ContrastiveLoss_43877385896505
// MI455X (gfx1250) — matrix operand order repaired and checked by reading, not compiled, not run on hardware
//
#include <hip/hip_runtime.h>
#include <hip/hip_bf16.h>
#include <math.h>

// ---------------------------------------------------------------------------
// ContrastiveLoss on MI455X (gfx1250, wave32, WMMA).
// sim = N N^T via bf16x2-split V_WMMA_F32_16X16X32_BF16 (3 wmma per K=32),
// fused exp/mask/row-reduction epilogue, final scalar reduction.
// ---------------------------------------------------------------------------

typedef __attribute__((ext_vector_type(16))) __bf16 bf16x16;
typedef __attribute__((ext_vector_type(8)))  __bf16 bf16x8;
typedef __attribute__((ext_vector_type(8)))  float  f32x8;

#define DIM 256          // embedding dimension D
#define ROWS_PER_BLOCK 64

// ------------------------- kernel 1: normalize + bf16 hi/lo split ----------
__global__ __launch_bounds__(DIM)
void cl_normalize_kernel(const float* __restrict__ emb,
                         __bf16* __restrict__ ahi,
                         __bf16* __restrict__ alo)
{
    const int row = blockIdx.x;
    const int t   = threadIdx.x;            // 256 threads == D
    float x = emb[(size_t)row * DIM + t];

    __shared__ float red[DIM];
    red[t] = x * x;
    __syncthreads();
    #pragma unroll
    for (int s = DIM / 2; s > 0; s >>= 1) {
        if (t < s) red[t] += red[t + s];
        __syncthreads();
    }
    const float inv = 1.0f / sqrtf(red[0]);
    const float v   = x * inv;
    const __bf16 hi = (__bf16)v;
    const __bf16 lo = (__bf16)(v - (float)hi);
    ahi[(size_t)row * DIM + t] = hi;
    alo[(size_t)row * DIM + t] = lo;
}

// ------------------------- kernel 2: fused sim/exp/mask/row-sums -----------
__global__ __launch_bounds__(256)
void cl_sim_kernel(const __bf16* __restrict__ Ahi,
                   const __bf16* __restrict__ Alo,
                   const float*  __restrict__ vel,
                   const float*  __restrict__ pos2,   // [B][2]
                   const float*  __restrict__ nq,
                   const float*  __restrict__ cpu,
                   float* __restrict__ pos_sum,
                   float* __restrict__ neg_sum,
                   int B)
{
    __shared__ float lds_pos[ROWS_PER_BLOCK];
    __shared__ float lds_neg[ROWS_PER_BLOCK];

    const int t = threadIdx.x;
    if (t < ROWS_PER_BLOCK) { lds_pos[t] = 0.0f; lds_neg[t] = 0.0f; }
    __syncthreads();

    const int w      = t >> 5;          // wave id 0..7
    const int lane   = t & 31;
    const int rt     = w & 3;           // row tile within 64-row strip
    const int cphase = w >> 2;          // 0/1: interleaved column tiles
    const int h      = lane >> 4;       // half-wave
    const int l      = lane & 15;
    const int rowbase = blockIdx.x * ROWS_PER_BLOCK + rt * 16;

    // Row features for the 8 rows this lane's C elements map to (M = r + 8h).
    float vi[8], pxi[8], pyi[8], nqi[8], cpi[8];
    #pragma unroll
    for (int r = 0; r < 8; ++r) {
        const int i = rowbase + 8 * h + r;
        vi[r]  = vel[i];
        pxi[r] = pos2[2 * i];
        pyi[r] = pos2[2 * i + 1];
        nqi[r] = nq[i];
        cpi[r] = cpu[i];
    }

    float accp[8] = {0,0,0,0,0,0,0,0};
    float accn[8] = {0,0,0,0,0,0,0,0};

    // A-fragment source row for this lane (A layout: M = lane&15).
    const __bf16* arow_hi = Ahi + (size_t)(rowbase + l) * DIM;
    const __bf16* arow_lo = Alo + (size_t)(rowbase + l) * DIM;

    const int nct = B >> 4;             // column tiles of 16
    for (int ct = cphase; ct < nct; ct += 2) {
        const int colbase = ct * 16;
        const int j = colbase + l;      // this lane's output column

        const float vj  = vel[j];
        const float pxj = pos2[2 * j];
        const float pyj = pos2[2 * j + 1];
        const float nqj = nq[j];
        const float cpj = cpu[j];

        // B-fragment: column j of sim == row j of normalized embeddings.
        const __bf16* brow_hi = Ahi + (size_t)j * DIM + 8 * h;
        const __bf16* brow_lo = Alo + (size_t)j * DIM + 8 * h;

        f32x8 c = {};
        #pragma unroll
        for (int kk = 0; kk < DIM; kk += 32) {
            // A 16x32 bf16 frag: lane (m=l, half h) holds K=8h..8h+7, 16+8h..+7
            bf16x8 ah0 = *(const bf16x8*)(arow_hi + kk + 8 * h);
            bf16x8 ah1 = *(const bf16x8*)(arow_hi + kk + 8 * h + 16);
            bf16x8 al0 = *(const bf16x8*)(arow_lo + kk + 8 * h);
            bf16x8 al1 = *(const bf16x8*)(arow_lo + kk + 8 * h + 16);
            bf16x16 ah = __builtin_shufflevector(ah0, ah1,
                0,1,2,3,4,5,6,7,8,9,10,11,12,13,14,15);
            bf16x16 al = __builtin_shufflevector(al0, al1,
                0,1,2,3,4,5,6,7,8,9,10,11,12,13,14,15);
            // B 32x16 bf16 frag: lane (n=l, half h) holds K=8h..8h+7, 16+8h..+7 of col n
            bf16x8 bh0 = *(const bf16x8*)(brow_hi + kk);
            bf16x8 bh1 = *(const bf16x8*)(brow_hi + kk + 16);
            bf16x8 bl0 = *(const bf16x8*)(brow_lo + kk);
            bf16x8 bl1 = *(const bf16x8*)(brow_lo + kk + 16);
            bf16x16 bh = __builtin_shufflevector(bh0, bh1,
                0,1,2,3,4,5,6,7,8,9,10,11,12,13,14,15);
            bf16x16 bl = __builtin_shufflevector(bl0, bl1,
                0,1,2,3,4,5,6,7,8,9,10,11,12,13,14,15);

            // fp32-ish product: Ahi*Bhi + Ahi*Blo + Alo*Bhi
            c = __builtin_amdgcn_wmma_f32_16x16x32_bf16(
                    false, ah, false, bh, (short)0, c, false, false);
            c = __builtin_amdgcn_wmma_f32_16x16x32_bf16(
                    false, ah, false, bl, (short)0, c, false, false);
            c = __builtin_amdgcn_wmma_f32_16x16x32_bf16(
                    false, al, false, bh, (short)0, c, false, false);
        }

        // Fused epilogue: logits = sim/0.1, masks, row partial sums.
        #pragma unroll
        for (int r = 0; r < 8; ++r) {
            const int i = rowbase + 8 * h + r;
            const float s = c[r];
            const float e = expf(s * 10.0f);
            const bool neq = (i != j);

            const float dv  = fabsf(vi[r] - vj);
            const float den = fmaxf(fabsf(vi[r]) + fabsf(vj) + 1e-8f, 1.0f);
            bool ok = (1.0f - dv / den) >= 0.8f;
            const float dx = pxi[r] - pxj;
            const float dy = pyi[r] - pyj;
            ok = ok && (dx * dx + dy * dy) <= 2500.0f;       // dist <= 50
            ok = ok && (1.0f - fabsf(nqi[r] - nqj)) >= 0.8f;
            ok = ok && (1.0f - fabsf(cpi[r] - cpj)) >= 0.8f;
            ok = ok && neq;

            accn[r] += neq ? e : 0.0f;
            accp[r] += ok  ? e : 0.0f;
        }
    }

    // Reduce per-row partials across lanes/waves via LDS float atomics.
    #pragma unroll
    for (int r = 0; r < 8; ++r) {
        const int rloc = rt * 16 + 8 * h + r;
        atomicAdd(&lds_pos[rloc], accp[r]);
        atomicAdd(&lds_neg[rloc], accn[r]);
    }
    __syncthreads();
    if (t < ROWS_PER_BLOCK) {
        pos_sum[blockIdx.x * ROWS_PER_BLOCK + t] = lds_pos[t];
        neg_sum[blockIdx.x * ROWS_PER_BLOCK + t] = lds_neg[t];
    }
}

// ------------------------- kernel 3: final scalar reduction ----------------
__global__ __launch_bounds__(256)
void cl_finalize_kernel(const float* __restrict__ pos_sum,
                        const float* __restrict__ neg_sum,
                        float* __restrict__ out, int B)
{
    __shared__ float sl[256];
    __shared__ float sc[256];
    const int t = threadIdx.x;
    float loss = 0.0f, cnt = 0.0f;
    for (int i = t; i < B; i += 256) {
        const float p = pos_sum[i];
        const float n = neg_sum[i];
        if (p > 0.0f) {                   // has_pos (exp > 0 always)
            loss += -logf(p / (p + n));
            cnt  += 1.0f;
        }
    }
    sl[t] = loss; sc[t] = cnt;
    __syncthreads();
    #pragma unroll
    for (int s = 128; s > 0; s >>= 1) {
        if (t < s) { sl[t] += sl[t + s]; sc[t] += sc[t + s]; }
        __syncthreads();
    }
    if (t == 0) out[0] = (sc[0] > 0.0f) ? (sl[0] / sc[0]) : 0.0f;
}

// ------------------------- launcher ----------------------------------------
extern "C" void kernel_launch(void* const* d_in, const int* in_sizes, int n_in,
                              void* d_out, int out_size, void* d_ws, size_t ws_size,
                              hipStream_t stream)
{
    const float* emb  = (const float*)d_in[0];   // [B, D]
    const float* vel  = (const float*)d_in[1];   // [B]
    const float* pos2 = (const float*)d_in[2];   // [B, 2]
    const float* nq   = (const float*)d_in[3];   // [B]
    const float* cpu  = (const float*)d_in[4];   // [B]

    const int B = in_sizes[1];                   // 8192
    // D fixed at 256 (DIM) per reference.

    char* ws = (char*)d_ws;
    __bf16* Ahi = (__bf16*)ws;                                   // B*D bf16
    __bf16* Alo = (__bf16*)(ws + (size_t)B * DIM * 2);           // B*D bf16
    float*  psum = (float*)(ws + (size_t)B * DIM * 4);           // B floats
    float*  nsum = psum + B;                                     // B floats

    cl_normalize_kernel<<<B, DIM, 0, stream>>>(emb, Ahi, Alo);
    cl_sim_kernel<<<B / ROWS_PER_BLOCK, 256, 0, stream>>>(
        Ahi, Alo, vel, pos2, nq, cpu, psum, nsum, B);
    cl_finalize_kernel<<<1, 256, 0, stream>>>(psum, nsum, (float*)d_out, B);
}
